// raw_gru_adde_60971355734182
// MI455X (gfx1250) — matrix operand order repaired and checked by reading, not compiled, not run on hardware
//
#include <hip/hip_runtime.h>

// ---------------------------------------------------------------------------
// CDNA5 (gfx1250) implementation of the GAT+GRU reference.
// All heavy GEMMs use v_wmma_f32_16x16x32_bf16 (wave32, 16x16 C tiles,
// one wave computes a 16x128 output strip; B pre-packed into fragment layout).
// ---------------------------------------------------------------------------

typedef __attribute__((ext_vector_type(16))) __bf16 v16bf;
typedef __attribute__((ext_vector_type(8)))  float  v8f;

#define NN   20000
#define EEDG 160000
#define DDIM 128
#define HHD  4
#define CCV  3
#define GGR  512
#define INF_ 74
#define EFD  11

__device__ __forceinline__ unsigned short f2bf(float f) {
  unsigned int u = __float_as_uint(f);
  u += 0x7fffu + ((u >> 16) & 1u);          // round-to-nearest-even
  return (unsigned short)(u >> 16);
}
__device__ __forceinline__ float bf2f(unsigned short s) {
  return __uint_as_float(((unsigned int)s) << 16);
}
__device__ __forceinline__ float wave_allreduce(float v) {
#pragma unroll
  for (int off = 16; off >= 1; off >>= 1) v += __shfl_xor(v, off, 32);
  return v;
}

// ---------------------------------------------------------------------------
// WMMA GEMM:  D[M,128] = A[M,Kp](bf16,row-major,lda) @ Bpacked[Kp,128](bf16)
// Bpacked layout: tile t=(kt,nt) of 32x16; lane l holds col n=nt*16+(l&15),
// K = kt*32 + (l>>4)*16 + e   (e = 0..15), stored contiguously: t*512+l*16+e.
// A fragment (matches B): lane<16 -> K 0..15; lane>=16 -> K 16..31.
// Epilogue: optional bias, f32 store / bf16 store / f32 scatter-atomic-add.
// ---------------------------------------------------------------------------
__global__ void wmma_gemm_bf16(const unsigned short* __restrict__ A, int lda,
                               int Kp, int M,
                               const unsigned short* __restrict__ Bp,
                               float* __restrict__ Df,
                               unsigned short* __restrict__ Db,
                               int dstride, int dbase,
                               const float* __restrict__ bias,
                               const int* __restrict__ scatter) {
  const int waveInBlock = threadIdx.x >> 5;
  const int lane        = threadIdx.x & 31;
  const int tileM = (blockIdx.x * 8 + waveInBlock) * 16;
  if (tileM >= M) return;                        // wave-uniform exit

  const int halfSel = lane >> 4;                 // 0 or 1
  int rowA = tileM + (lane & 15);
  if (rowA >= M) rowA = M - 1;                   // clamp (results masked at store)
  const unsigned short* arow = A + (size_t)rowA * lda + halfSel * 16;

  v8f acc[8];
#pragma unroll
  for (int nt = 0; nt < 8; ++nt)
#pragma unroll
    for (int e = 0; e < 8; ++e) acc[nt][e] = 0.0f;

  union Frag { v16bf v; uint4 u[2]; };
  const unsigned short* bl = Bp + lane * 16;

  const int nk = Kp >> 5;
  for (int kt = 0; kt < nk; ++kt) {
    Frag af;
    const unsigned short* ap = arow + kt * 32;
    af.u[0] = *(const uint4*)(ap);
    af.u[1] = *(const uint4*)(ap + 8);
    const unsigned short* bt = bl + kt * 8 * 512;
#pragma unroll
    for (int nt = 0; nt < 8; ++nt) {
      Frag bf;
      bf.u[0] = *(const uint4*)(bt + nt * 512);
      bf.u[1] = *(const uint4*)(bt + nt * 512 + 8);
      acc[nt] = __builtin_amdgcn_wmma_f32_16x16x32_bf16(
          false, af.v, false, bf.v, (short)0, acc[nt], false, false);
    }
  }

  const int mbase = tileM + halfSel * 8;
  const int colb  = lane & 15;
#pragma unroll
  for (int nt = 0; nt < 8; ++nt) {
    const int col  = nt * 16 + colb;
    const float bv = bias ? bias[col] : 0.0f;
#pragma unroll
    for (int r = 0; r < 8; ++r) {
      const int m = mbase + r;
      if (m >= M) continue;
      float val = acc[nt][r] + bv;
      if (scatter) {
        unsafeAtomicAdd(&Df[(size_t)scatter[m] * dstride + col], val);
      } else {
        size_t o = (size_t)m * dstride + dbase + col;
        if (Df) Df[o] = val;
        if (Db) Db[o] = f2bf(val);
      }
    }
  }
}

// Pack f32 weight [K,128] (row stride 128) -> bf16 fragment layout, K padded to Kp.
__global__ void pack_b_kernel(const float* __restrict__ W, int K, int Kp,
                              unsigned short* __restrict__ out) {
  int idx = blockIdx.x * blockDim.x + threadIdx.x;
  if (idx >= Kp * 128) return;
  int t = idx >> 9, rem = idx & 511;
  int lane = rem >> 4, e = rem & 15;
  int kt = t >> 3, nt = t & 7;
  int k = kt * 32 + (lane >> 4) * 16 + e;
  int n = nt * 16 + (lane & 15);
  float v = (k < K) ? W[(size_t)k * 128 + n] : 0.0f;
  out[idx] = f2bf(v);
}

// M1 = Wq @ Wk^T  (both [128,128] row-major)
__global__ void m1_kernel(const float* __restrict__ Wq,
                          const float* __restrict__ Wk,
                          float* __restrict__ M1) {
  int idx = blockIdx.x * blockDim.x + threadIdx.x;
  if (idx >= 128 * 128) return;
  int a = idx >> 7, b = idx & 127;
  float s = 0.0f;
  for (int f = 0; f < 128; ++f) s += Wq[a * 128 + f] * Wk[b * 128 + f];
  M1[idx] = s;
}

// f32 [rows,srcCols] -> bf16 [rows,dstCols] zero-padded
__global__ void cvt_pad_bf16(const float* __restrict__ src, int srcCols,
                             unsigned short* __restrict__ dst, int dstCols,
                             int rows) {
  size_t idx = (size_t)blockIdx.x * blockDim.x + threadIdx.x;
  if (idx >= (size_t)rows * dstCols) return;
  int r = (int)(idx / dstCols), c = (int)(idx % dstCols);
  float v = (c < srcCols) ? src[(size_t)r * srcCols + c] : 0.0f;
  dst[idx] = f2bf(v);
}

// x[3e] = z[src[e]], x[3e+1] = z[dst[e]]  (16B chunks)
__global__ void gather_x(const unsigned short* __restrict__ z,
                         const int* __restrict__ src,
                         const int* __restrict__ dst,
                         unsigned short* __restrict__ x, int E) {
  int idx = blockIdx.x * blockDim.x + threadIdx.x;
  if (idx >= E * 16) return;
  int e = idx >> 4, c = (idx & 15) * 8;
  *(uint4*)(x + (size_t)(3 * e) * 128 + c) =
      *(const uint4*)(z + (size_t)src[e] * 128 + c);
  *(uint4*)(x + (size_t)(3 * e + 1) * 128 + c) =
      *(const uint4*)(z + (size_t)dst[e] * 128 + c);
}

__global__ void count_kernel(const int* __restrict__ ids,
                             float* __restrict__ cnt, int n) {
  int i = blockIdx.x * blockDim.x + threadIdx.x;
  if (i < n) unsafeAtomicAdd(&cnt[ids[i]], 1.0f);
}

__global__ void g74_accum(const float* __restrict__ h0,
                          const int* __restrict__ gid,
                          float* __restrict__ g74, int N) {
  int idx = blockIdx.x * blockDim.x + threadIdx.x;
  if (idx >= N * INF_) return;
  int n = idx / INF_, d = idx % INF_;
  unsafeAtomicAdd(&g74[(size_t)gid[n] * 80 + d], h0[idx]);
}

__global__ void x0_kernel(const float* __restrict__ g74,
                          const float* __restrict__ gcount,
                          const float* __restrict__ projW,
                          const float* __restrict__ projB,
                          float* __restrict__ seq0) {
  int idx = blockIdx.x * blockDim.x + threadIdx.x;
  if (idx >= GGR * 128) return;
  int g = idx >> 7, f = idx & 127;
  float inv = 1.0f / fmaxf(gcount[g], 1.0f);
  float acc = projB[f];
  for (int d = 0; d < INF_; ++d)
    acc += g74[(size_t)g * 80 + d] * inv * projW[d * 128 + f];
  seq0[idx] = acc;
}

// per-edge 3x3 attention (one wave per edge): S_ij = T_i . x_j ; u = sum_j w_j x_j
__global__ void attn_kernel(const unsigned short* __restrict__ x,
                            const unsigned short* __restrict__ T,
                            unsigned short* __restrict__ u, int E) {
  int wave = (int)(((size_t)blockIdx.x * blockDim.x + threadIdx.x) >> 5);
  int lane = threadIdx.x & 31;
  if (wave >= E) return;
  size_t base = (size_t)wave * 3 * 128;
  float xv[3][4], tv[3][4];
#pragma unroll
  for (int j = 0; j < 3; ++j) {
    uint2 px = *(const uint2*)(x + base + j * 128 + lane * 4);
    xv[j][0] = bf2f((unsigned short)(px.x & 0xffff));
    xv[j][1] = bf2f((unsigned short)(px.x >> 16));
    xv[j][2] = bf2f((unsigned short)(px.y & 0xffff));
    xv[j][3] = bf2f((unsigned short)(px.y >> 16));
    uint2 pt = *(const uint2*)(T + base + j * 128 + lane * 4);
    tv[j][0] = bf2f((unsigned short)(pt.x & 0xffff));
    tv[j][1] = bf2f((unsigned short)(pt.x >> 16));
    tv[j][2] = bf2f((unsigned short)(pt.y & 0xffff));
    tv[j][3] = bf2f((unsigned short)(pt.y >> 16));
  }
  float s[3][3];
#pragma unroll
  for (int i = 0; i < 3; ++i)
#pragma unroll
    for (int j = 0; j < 3; ++j) {
      float p = 0.0f;
#pragma unroll
      for (int c = 0; c < 4; ++c) p += tv[i][c] * xv[j][c];
      s[i][j] = wave_allreduce(p);
    }
  float w[3] = {0.0f, 0.0f, 0.0f};
#pragma unroll
  for (int i = 0; i < 3; ++i) {
    float mx = fmaxf(s[i][0], fmaxf(s[i][1], s[i][2]));
    float e0 = expf(s[i][0] - mx), e1 = expf(s[i][1] - mx), e2 = expf(s[i][2] - mx);
    float inv = 1.0f / (e0 + e1 + e2);
    w[0] += e0 * inv; w[1] += e1 * inv; w[2] += e2 * inv;
  }
  float uv[4];
#pragma unroll
  for (int c = 0; c < 4; ++c)
    uv[c] = w[0] * xv[0][c] + w[1] * xv[1][c] + w[2] * xv[2][c];
  uint2 o;
  o.x = (unsigned int)f2bf(uv[0]) | ((unsigned int)f2bf(uv[1]) << 16);
  o.y = (unsigned int)f2bf(uv[2]) | ((unsigned int)f2bf(uv[3]) << 16);
  *(uint2*)(u + (size_t)wave * 128 + lane * 4) = o;
}

// one wave per node: relu(hsum/deg) -> layernorm -> hc[:, head*128:+128] (bf16)
__global__ void finalize_head(const float* __restrict__ hsum,
                              const float* __restrict__ deg,
                              const float* __restrict__ g,
                              const float* __restrict__ b,
                              unsigned short* __restrict__ hc,
                              int head, int N) {
  int wave = (int)(((size_t)blockIdx.x * blockDim.x + threadIdx.x) >> 5);
  int lane = threadIdx.x & 31;
  if (wave >= N) return;
  float inv = 1.0f / fmaxf(deg[wave], 1.0f);
  float v[4], s = 0.0f;
#pragma unroll
  for (int c = 0; c < 4; ++c) {
    float t = hsum[(size_t)wave * 128 + lane * 4 + c] * inv;
    v[c] = fmaxf(t, 0.0f);
    s += v[c];
  }
  float mean = wave_allreduce(s) * (1.0f / 128.0f);
  float q = 0.0f;
#pragma unroll
  for (int c = 0; c < 4; ++c) { float d = v[c] - mean; q += d * d; }
  float var = wave_allreduce(q) * (1.0f / 128.0f);
  float rstd = rsqrtf(var + 1e-5f);
#pragma unroll
  for (int c = 0; c < 4; ++c) {
    int d = lane * 4 + c;
    hc[(size_t)wave * 512 + head * 128 + d] =
        f2bf((v[c] - mean) * rstd * g[d] + b[d]);
  }
}

__global__ void seg_accum_bf(const unsigned short* __restrict__ h,
                             const int* __restrict__ gid,
                             float* __restrict__ gsum, int N) {
  size_t idx = (size_t)blockIdx.x * blockDim.x + threadIdx.x;
  if (idx >= (size_t)N * 128) return;
  int n = (int)(idx >> 7);
  unsafeAtomicAdd(&gsum[(size_t)gid[n] * 128 + (idx & 127)], bf2f(h[idx]));
}

__global__ void seg_div(const float* __restrict__ gsum,
                        const float* __restrict__ gcount,
                        float* __restrict__ out) {
  int idx = blockIdx.x * blockDim.x + threadIdx.x;
  if (idx >= GGR * 128) return;
  out[idx] = gsum[idx] / fmaxf(gcount[idx >> 7], 1.0f);
}

// one GRU step (torch gate order r,z,n); thread per (batch,dim)
__global__ void gru_step(const float* __restrict__ x, int in_dim,
                         const float* __restrict__ h_prev,
                         float* __restrict__ h_out,
                         const float* __restrict__ Wih,
                         const float* __restrict__ Whh,
                         const float* __restrict__ bih,
                         const float* __restrict__ bhh, int B) {
  int idx = blockIdx.x * blockDim.x + threadIdx.x;
  if (idx >= B * 128) return;
  int b = idx >> 7, d = idx & 127;
  const float* xb = x + (size_t)b * in_dim;
  const float* hb = h_prev + (size_t)b * 128;
  float ir = bih[d], iz = bih[128 + d], in_ = bih[256 + d];
  for (int k = 0; k < in_dim; ++k) {
    float xv = xb[k];
    const float* wr = Wih + (size_t)k * 384;
    ir += xv * wr[d]; iz += xv * wr[128 + d]; in_ += xv * wr[256 + d];
  }
  float hr = bhh[d], hz = bhh[128 + d], hn = bhh[256 + d];
  for (int k = 0; k < 128; ++k) {
    float hv = hb[k];
    const float* wr = Whh + (size_t)k * 384;
    hr += hv * wr[d]; hz += hv * wr[128 + d]; hn += hv * wr[256 + d];
  }
  float r = 1.0f / (1.0f + expf(-(ir + hr)));
  float z = 1.0f / (1.0f + expf(-(iz + hz)));
  float n = tanhf(in_ + r * hn);
  h_out[idx] = (1.0f - z) * n + z * hb[d];
}

__global__ void concat_y(const float* __restrict__ yf,
                         const float* __restrict__ yb,
                         float* __restrict__ y1in) {
  int idx = blockIdx.x * blockDim.x + threadIdx.x;
  if (idx >= 4 * GGR * 256) return;
  int t = idx / (GGR * 256), rem = idx % (GGR * 256);
  int b = rem / 256, c = rem % 256;
  y1in[idx] = (c < 128) ? yf[((size_t)t * GGR + b) * 128 + c]
                        : yb[((size_t)t * GGR + b) * 128 + (c - 128)];
}

__global__ void final_sum(const float* __restrict__ y0f,
                          const float* __restrict__ y0b,
                          const float* __restrict__ y1f,
                          const float* __restrict__ y1b,
                          float* __restrict__ out) {
  int idx = blockIdx.x * blockDim.x + threadIdx.x;
  if (idx >= GGR * 128) return;
  const int last = 3 * GGR * 128;
  out[idx] = y0f[last + idx] + y0b[idx] + y1f[last + idx] + y1b[idx];
}

// ---------------------------------------------------------------------------
extern "C" void kernel_launch(void* const* d_in, const int* in_sizes, int n_in,
                              void* d_out, int out_size, void* d_ws,
                              size_t ws_size, hipStream_t stream) {
  (void)in_sizes; (void)n_in; (void)out_size; (void)ws_size;
  const int N = NN, E = EEDG, D = DDIM, H = HHD, C = CCV, G = GGR;

  const float* h0     = (const float*)d_in[0];
  const float* e_f    = (const float*)d_in[1];
  const int*   src    = (const int*)d_in[2];
  const int*   dst    = (const int*)d_in[3];
  const int*   gid    = (const int*)d_in[4];
  const float* lin0_W = (const float*)d_in[6];
  const float* lin_W  = (const float*)d_in[7];
  const float* We_W   = (const float*)d_in[8];
  const float* Wq_W   = (const float*)d_in[9];
  const float* Wk_W   = (const float*)d_in[10];
  const float* Wv_W   = (const float*)d_in[11];
  const float* ln_g   = (const float*)d_in[12];
  const float* ln_b   = (const float*)d_in[13];
  const float* out_W  = (const float*)d_in[14];
  const float* out_b  = (const float*)d_in[15];
  const float* proj_W = (const float*)d_in[16];
  const float* proj_b = (const float*)d_in[17];
  const float* Wih0   = (const float*)d_in[18];
  const float* Whh0   = (const float*)d_in[19];
  const float* bih0   = (const float*)d_in[20];
  const float* bhh0   = (const float*)d_in[21];
  const float* Wih1   = (const float*)d_in[22];
  const float* Whh1   = (const float*)d_in[23];
  const float* bih1   = (const float*)d_in[24];
  const float* bhh1   = (const float*)d_in[25];
  float* out = (float*)d_out;

  // ---- workspace carve (256B aligned) ----
  char* w = (char*)d_ws;
  size_t off = 0;
  auto alloc = [&](size_t bytes) -> void* {
    void* p = w + off;
    off = (off + bytes + 255) & ~(size_t)255;
    return p;
  };
  float* deg     = (float*)alloc((size_t)N * 4);
  float* gcount  = (float*)alloc((size_t)G * 4);
  float* g74     = (float*)alloc((size_t)G * 80 * 4);
  float* zerobuf = (float*)alloc((size_t)G * 128 * 4);
  float* M1tmp   = (float*)alloc(128 * 128 * 4);
  float* gsum    = (float*)alloc((size_t)G * 128 * 4);
  float* hsum    = (float*)alloc((size_t)N * 128 * 4);
  float* seq     = (float*)alloc((size_t)4 * G * 128 * 4);
  float* y0f     = (float*)alloc((size_t)4 * G * 128 * 4);
  float* y0b     = (float*)alloc((size_t)4 * G * 128 * 4);
  float* y1f     = (float*)alloc((size_t)4 * G * 128 * 4);
  float* y1b     = (float*)alloc((size_t)4 * G * 128 * 4);
  float* y1in    = (float*)alloc((size_t)4 * G * 256 * 4);
  unsigned short* h_bf  = (unsigned short*)alloc((size_t)N * 128 * 2);
  unsigned short* z_bf  = (unsigned short*)alloc((size_t)N * 128 * 2);
  unsigned short* hc_bf = (unsigned short*)alloc((size_t)N * 512 * 2);
  unsigned short* ef_bf = (unsigned short*)alloc((size_t)E * 32 * 2);
  unsigned short* x_bf  = (unsigned short*)alloc((size_t)3 * E * 128 * 2);
  unsigned short* T_bf  = (unsigned short*)alloc((size_t)3 * E * 128 * 2);
  unsigned short* u_bf  = (unsigned short*)alloc((size_t)E * 128 * 2);
  unsigned short* pWl   = (unsigned short*)alloc((size_t)C * H * 16384 * 2);
  unsigned short* pWe   = (unsigned short*)alloc((size_t)C * H * 4096 * 2);
  unsigned short* pM1   = (unsigned short*)alloc((size_t)C * H * 16384 * 2);
  unsigned short* pWv   = (unsigned short*)alloc((size_t)C * H * 16384 * 2);
  unsigned short* pOut  = (unsigned short*)alloc((size_t)C * 65536 * 2);

  auto gemm = [&](const unsigned short* A, int lda, int Kp, int M,
                  const unsigned short* Bp, float* Df, unsigned short* Db,
                  int dstride, int dbase, const float* bias,
                  const int* scatter) {
    int grid = (M + 127) / 128;
    wmma_gemm_bf16<<<grid, 256, 0, stream>>>(A, lda, Kp, M, Bp, Df, Db,
                                             dstride, dbase, bias, scatter);
  };
  auto pack = [&](const float* W, int K, int Kp, unsigned short* dstp) {
    int total = Kp * 128;
    pack_b_kernel<<<(total + 255) / 256, 256, 0, stream>>>(W, K, Kp, dstp);
  };

  // ---- init / counts / conversions ----
  hipMemsetAsync(deg, 0, (size_t)N * 4, stream);
  hipMemsetAsync(gcount, 0, (size_t)G * 4, stream);
  hipMemsetAsync(g74, 0, (size_t)G * 80 * 4, stream);
  hipMemsetAsync(zerobuf, 0, (size_t)G * 128 * 4, stream);
  count_kernel<<<(E + 255) / 256, 256, 0, stream>>>(dst, deg, E);
  count_kernel<<<(N + 255) / 256, 256, 0, stream>>>(gid, gcount, N);
  g74_accum<<<(N * INF_ + 255) / 256, 256, 0, stream>>>(h0, gid, g74, N);
  cvt_pad_bf16<<<((size_t)N * 128 + 255) / 256, 256, 0, stream>>>(h0, INF_, h_bf, 128, N);
  cvt_pad_bf16<<<((size_t)E * 32 + 255) / 256, 256, 0, stream>>>(e_f, EFD, ef_bf, 32, E);
  x0_kernel<<<(G * 128 + 255) / 256, 256, 0, stream>>>(g74, gcount, proj_W, proj_b, seq);

  // ---- pack all weights (tiny; once per launch) ----
  for (int j = 0; j < C; ++j)
    for (int i = 0; i < H; ++i) {
      int s = j * H + i;
      const float* Wl = (j == 0) ? (lin0_W + (size_t)i * INF_ * D)
                                 : (lin_W + (size_t)((j - 1) * H + i) * D * D);
      pack(Wl, (j == 0) ? INF_ : D, 128, pWl + (size_t)s * 16384);
      pack(We_W + (size_t)s * EFD * D, EFD, 32, pWe + (size_t)s * 4096);
      m1_kernel<<<(16384 + 255) / 256, 256, 0, stream>>>(
          Wq_W + (size_t)s * D * D, Wk_W + (size_t)s * D * D, M1tmp);
      pack(M1tmp, 128, 128, pM1 + (size_t)s * 16384);
      pack(Wv_W + (size_t)s * D * D, 128, 128, pWv + (size_t)s * 16384);
    }
  for (int j = 0; j < C; ++j)
    pack(out_W + (size_t)j * 512 * 128, 512, 512, pOut + (size_t)j * 65536);

  // ---- GNN layers ----
  for (int j = 0; j < C; ++j) {
    for (int i = 0; i < H; ++i) {
      int s = j * H + i;
      // z = h @ Wl        [N,128] bf16
      gemm(h_bf, 128, 128, N, pWl + (size_t)s * 16384, nullptr, z_bf, 128, 0,
           nullptr, nullptr);
      // ef proj -> x rows 3e+2 (row stride 384, base 256)
      gemm(ef_bf, 32, 32, E, pWe + (size_t)s * 4096, nullptr, x_bf, 384, 256,
           nullptr, nullptr);
      // x rows 3e, 3e+1 from z[src], z[dst]
      gather_x<<<(E * 16 + 255) / 256, 256, 0, stream>>>(z_bf, src, dst, x_bf, E);
      // T = x @ (Wq Wk^T)  [3E,128]
      gemm(x_bf, 128, 128, 3 * E, pM1 + (size_t)s * 16384, nullptr, T_bf, 128,
           0, nullptr, nullptr);
      hipMemsetAsync(hsum, 0, (size_t)N * 128 * 4, stream);
      // per-edge softmax: u = sum_j w_j x_j
      attn_kernel<<<(E + 7) / 8, 256, 0, stream>>>(x_bf, T_bf, u_bf, E);
      // m = u @ Wv, scatter-add into hsum[dst]
      gemm(u_bf, 128, 128, E, pWv + (size_t)s * 16384, hsum, nullptr, 128, 0,
           nullptr, dst);
      // relu + mean + layernorm -> concat slot
      finalize_head<<<(N + 7) / 8, 256, 0, stream>>>(
          hsum, deg, ln_g + (size_t)s * 128, ln_b + (size_t)s * 128, hc_bf, i, N);
    }
    // h = concat(heads) @ out_W + out_b  -> bf16 h for next layer
    gemm(hc_bf, 512, 512, N, pOut + (size_t)j * 65536, nullptr, h_bf, 128, 0,
         out_b + (size_t)j * 128, nullptr);
    hipMemsetAsync(gsum, 0, (size_t)G * 128 * 4, stream);
    seg_accum_bf<<<((size_t)N * 128 + 255) / 256, 256, 0, stream>>>(h_bf, gid, gsum, N);
    seg_div<<<(G * 128 + 255) / 256, 256, 0, stream>>>(
        gsum, gcount, seq + (size_t)(1 + j) * G * 128);
  }

  // ---- bidirectional 2-layer GRU over seq[4,G,128] ----
  int gblk = (G * 128 + 255) / 256;
  for (int t = 0; t < 4; ++t) {
    const float* hp = (t == 0) ? zerobuf : (y0f + (size_t)(t - 1) * G * 128);
    gru_step<<<gblk, 256, 0, stream>>>(seq + (size_t)t * G * 128, 128, hp,
                                       y0f + (size_t)t * G * 128, Wih0, Whh0,
                                       bih0, bhh0, G);
  }
  for (int t = 3; t >= 0; --t) {
    const float* hp = (t == 3) ? zerobuf : (y0b + (size_t)(t + 1) * G * 128);
    gru_step<<<gblk, 256, 0, stream>>>(seq + (size_t)t * G * 128, 128, hp,
                                       y0b + (size_t)t * G * 128,
                                       Wih0 + 128 * 384, Whh0 + 128 * 384,
                                       bih0 + 384, bhh0 + 384, G);
  }
  concat_y<<<(4 * G * 256 + 255) / 256, 256, 0, stream>>>(y0f, y0b, y1in);
  for (int t = 0; t < 4; ++t) {
    const float* hp = (t == 0) ? zerobuf : (y1f + (size_t)(t - 1) * G * 128);
    gru_step<<<gblk, 256, 0, stream>>>(y1in + (size_t)t * G * 256, 256, hp,
                                       y1f + (size_t)t * G * 128, Wih1, Whh1,
                                       bih1, bhh1, G);
  }
  for (int t = 3; t >= 0; --t) {
    const float* hp = (t == 3) ? zerobuf : (y1b + (size_t)(t + 1) * G * 128);
    gru_step<<<gblk, 256, 0, stream>>>(y1in + (size_t)t * G * 256, 256, hp,
                                       y1b + (size_t)t * G * 128,
                                       Wih1 + 256 * 384, Whh1 + 128 * 384,
                                       bih1 + 384, bhh1 + 384, G);
  }
  final_sum<<<(G * 128 + 255) / 256, 256, 0, stream>>>(y0f, y0b, y1f, y1b, out);
}
